// EdgeGNN_69028714381394
// MI455X (gfx1250) — matrix operand order repaired and checked by reading, not compiled, not run on hardware
//
#include <hip/hip_runtime.h>

// ---------------------------------------------------------------------------
// EdgeGNN (GINE) forward for MI455X / gfx1250.
//  * GEMMs on v_wmma_f32_16x16x32_f16 with 3-term fp16 hi/lo split
//    (near-fp32 accuracy, fp32 accumulate).
//  * Weights pre-split to f16 hi/lo (transposed) once per GEMM; LDS staging is
//    pure b128 copies; fragments are contiguous ds_load_b128s (no VALU build).
//  * Previous layer's BN+ReLU fused into GEMM X-staging (xf) where the tensor
//    has a single GEMM consumer -> removes full streaming passes (t2: 820MB/layer).
//  * Training-mode BN stats fused into GEMM epilogue
//    (shfl_xor(16) -> ds_add_f32 -> one global atomic per column per block).
//  * Edge scatter (segment_sum) via global_atomic_add_f32 (unsafeAtomicAdd).
// ---------------------------------------------------------------------------

static constexpr int NN = 50000;    // nodes
static constexpr int NE = 800000;   // edges
static constexpr int NG = 64;       // graphs

typedef _Float16 half_t;
typedef __attribute__((ext_vector_type(16))) _Float16 v16h;
typedef __attribute__((ext_vector_type(8)))  _Float16 v8h;
typedef __attribute__((ext_vector_type(4)))  _Float16 v4h;
typedef __attribute__((ext_vector_type(8)))  float    v8f;

// ---------------------------------------------------------------- utilities
__global__ void zero_f32(float* __restrict__ p, long n) {
  long i = (long)blockIdx.x * blockDim.x + threadIdx.x;
  long st = (long)gridDim.x * blockDim.x;
  for (; i < n; i += st) p[i] = 0.f;
}

// Split W[K,N] fp32 -> transposed f16 hi/lo arrays [N,K].
__global__ void wsplit(const float* __restrict__ W, half_t* __restrict__ whi,
                       half_t* __restrict__ wlo, int K, int N) {
  const int i = blockIdx.x * 256 + threadIdx.x;
  if (i < K * N) {
    const int k = i / N, n = i % N;
    const float w = W[i];
    const half_t h = (half_t)w;
    whi[n * K + k] = h;
    wlo[n * K + k] = (half_t)(w - (float)h);
  }
}

// --------------------------------------------------------------- WMMA GEMM
// Y[M,N](cols +noff, ld=ldY) = act(X) @ W[K,N] + b ; optional fused input BN
// (act(x)=relu(x*xf[256+c]+xf[384+c]) when xf!=null) ; optional BN stats out.
// Block: 256 threads = 8 waves; each wave computes a 16-row x N strip.
template<int NT, int KT>
__global__ __launch_bounds__(256) void gemm_kernel(
    const float* __restrict__ X,
    const half_t* __restrict__ whi, const half_t* __restrict__ wlo,
    const float* __restrict__ bias, float* __restrict__ Y,
    float* __restrict__ gstats, const float* __restrict__ xf,
    int M, int ldY, int noff, int do_stats)
{
  constexpr int K = KT * 32, N = NT * 16, KP = K + 8;  // pad: 16B-aligned rows
  extern __shared__ char smemraw[];
  half_t* sXhi = (half_t*)smemraw;          // [128][KP]
  half_t* sXlo = sXhi + 128 * KP;           // [128][KP]
  half_t* sWhi = sXlo + 128 * KP;           // [N][KP]
  half_t* sWlo = sWhi + N * KP;             // [N][KP]
  float*  sS   = (float*)(sWlo + N * KP);   // [256] BN partials

  const int tid  = threadIdx.x;
  const int wave = tid >> 5;
  const int lane = tid & 31;
  const int m0   = blockIdx.x * 128;

  sS[tid] = 0.f;

  // Stage X tile: coalesced float4 load, optional BN+ReLU, split to hi/lo f16.
  for (int idx = tid; idx < 128 * (K / 4); idx += 256) {
    const int row = idx / (K / 4);
    const int c4  = (idx % (K / 4)) * 4;
    float v[4] = {0.f, 0.f, 0.f, 0.f};
    if (m0 + row < M) {
      const float4 g = *(const float4*)(X + (size_t)(m0 + row) * K + c4);
      v[0] = g.x; v[1] = g.y; v[2] = g.z; v[3] = g.w;
      if (xf) {
#pragma unroll
        for (int j = 0; j < 4; ++j) {
          const float t = v[j] * xf[256 + c4 + j] + xf[384 + c4 + j];
          v[j] = t > 0.f ? t : 0.f;
        }
      }
    }
    v4h hi, lo;
#pragma unroll
    for (int j = 0; j < 4; ++j) {
      const half_t h = (half_t)v[j];
      hi[j] = h;
      lo[j] = (half_t)(v[j] - (float)h);
    }
    *(v4h*)(sXhi + row * KP + c4) = hi;
    *(v4h*)(sXlo + row * KP + c4) = lo;
  }
  // Stage pre-split weights: straight 16B copies.
  for (int idx = tid; idx < N * (K / 8); idx += 256) {
    const int n  = idx / (K / 8);
    const int c8 = (idx % (K / 8)) * 8;
    *(v8h*)(sWhi + n * KP + c8) = *(const v8h*)(whi + (size_t)n * K + c8);
    *(v8h*)(sWlo + n * KP + c8) = *(const v8h*)(wlo + (size_t)n * K + c8);
  }
  __syncthreads();

  const int mn = lane & 15;   // A-row within wave tile / B-col / D-col
  const int hb = lane >> 4;   // half-wave selector
  v8f acc[NT] = {};

#pragma unroll
  for (int kt = 0; kt < KT; ++kt) {
    // A fragment 16x32 f16: per CDNA5 layout = two contiguous 8-half runs
    // (k = base+0..7 and base+16..23, base = kt*32 + hb*8).
    const int aoff = (wave * 16 + mn) * KP + kt * 32 + (hb ? 8 : 0);
    const v8h a0h = *(const v8h*)(sXhi + aoff);
    const v8h a1h = *(const v8h*)(sXhi + aoff + 16);
    const v8h a0l = *(const v8h*)(sXlo + aoff);
    const v8h a1l = *(const v8h*)(sXlo + aoff + 16);
    const v16h ahi = __builtin_shufflevector(a0h, a1h, 0,1,2,3,4,5,6,7,8,9,10,11,12,13,14,15);
    const v16h alo = __builtin_shufflevector(a0l, a1l, 0,1,2,3,4,5,6,7,8,9,10,11,12,13,14,15);

#pragma unroll
    for (int nt = 0; nt < NT; ++nt) {
      // B fragment 32x16 f16: two contiguous 8-half runs per lane
      // (col = mn, k = base+0..7 and base+16..23, base = kt*32 + hb*8).
      const int boff = (nt * 16 + mn) * KP + kt * 32 + (hb ? 8 : 0);
      const v8h b0h = *(const v8h*)(sWhi + boff);
      const v8h b1h = *(const v8h*)(sWhi + boff + 16);
      const v8h b0l = *(const v8h*)(sWlo + boff);
      const v8h b1l = *(const v8h*)(sWlo + boff + 16);
      const v16h bhi = __builtin_shufflevector(b0h, b1h, 0,1,2,3,4,5,6,7,8,9,10,11,12,13,14,15);
      const v16h blo = __builtin_shufflevector(b0l, b1l, 0,1,2,3,4,5,6,7,8,9,10,11,12,13,14,15);
      // 3-term split-fp16 product, fp32 accumulate (near-fp32 GEMM)
      acc[nt] = __builtin_amdgcn_wmma_f32_16x16x32_f16(false, ahi, false, bhi, (short)0, acc[nt], false, false);
      acc[nt] = __builtin_amdgcn_wmma_f32_16x16x32_f16(false, alo, false, bhi, (short)0, acc[nt], false, false);
      acc[nt] = __builtin_amdgcn_wmma_f32_16x16x32_f16(false, ahi, false, blo, (short)0, acc[nt], false, false);
    }
  }

  // Epilogue: bias, store, fused BN statistics.
#pragma unroll
  for (int nt = 0; nt < NT; ++nt) {
    const int n = nt * 16 + mn;
    const float b = bias[n];
    float s = 0.f, sq = 0.f;
#pragma unroll
    for (int r = 0; r < 8; ++r) {
      const int row = m0 + wave * 16 + r + (hb ? 8 : 0);
      const float v = acc[nt][r] + b;
      if (row < M) {
        Y[(size_t)row * ldY + noff + n] = v;
        s += v; sq += v * v;
      }
    }
    if (do_stats) {
      s  += __shfl_xor(s, 16, 32);    // lanes l & l+16 hold the same column
      sq += __shfl_xor(sq, 16, 32);
      if (lane < 16) {
        atomicAdd(&sS[n], s);         // ds_add_f32
        atomicAdd(&sS[128 + n], sq);
      }
    }
  }
  if (do_stats) {
    __syncthreads();
    if (tid < N) {
      unsafeAtomicAdd(&gstats[tid],       sS[tid]);        // global_atomic_add_f32
      unsafeAtomicAdd(&gstats[128 + tid], sS[128 + tid]);
    }
  }
}

// stats[0:128]=sum, [128:256]=sumsq -> [256:384]=scale, [384:512]=shift
__global__ void bn_finalize(float* __restrict__ st, const float* __restrict__ g,
                            const float* __restrict__ be, float invM, int N) {
  const int t = threadIdx.x;
  if (t < N) {
    const float mean = st[t] * invM;
    const float var  = st[128 + t] * invM - mean * mean;
    const float sc   = g[t] * rsqrtf(var + 1e-5f);
    st[256 + t] = sc;
    st[384 + t] = be[t] - mean * sc;
  }
}

__global__ void bn_apply(float* __restrict__ Y, const float* __restrict__ st,
                         long M, int N, int ldY) {
  const long tot = M * (long)(N / 4);
  long i = (long)blockIdx.x * blockDim.x + threadIdx.x;
  const long gs = (long)gridDim.x * blockDim.x;
  for (; i < tot; i += gs) {
    const long row = i / (N / 4);
    const int  c4  = (int)(i % (N / 4)) * 4;
    float* y = Y + row * ldY + c4;
#pragma unroll
    for (int j = 0; j < 4; ++j) {
      const float v = y[j] * st[256 + c4 + j] + st[384 + c4 + j];
      y[j] = v > 0.f ? v : 0.f;
    }
  }
}

// t1[e] = [ xl[src]+xl[dst] , e[e] ]   (E x 128)
__global__ void concat_gather(float* __restrict__ t1, const float* __restrict__ xl,
                              const float* __restrict__ e, const int* __restrict__ src,
                              const int* __restrict__ dst) {
  const long tot = (long)NE * 32;
  long i = (long)blockIdx.x * blockDim.x + threadIdx.x;
  const long gs = (long)gridDim.x * blockDim.x;
  for (; i < tot; i += gs) {
    const long ed = i >> 5;
    const int  c4 = (int)(i & 31) * 4;
    float4 o;
    if (c4 < 64) {
      const float4 a = *(const float4*)(xl + (size_t)src[ed] * 64 + c4);
      const float4 b = *(const float4*)(xl + (size_t)dst[ed] * 64 + c4);
      o = make_float4(a.x + b.x, a.y + b.y, a.z + b.z, a.w + b.w);
    } else {
      o = *(const float4*)(e + ed * 64 + (c4 - 64));
    }
    *(float4*)(t1 + ed * 128 + c4) = o;
  }
}

// aggr[dst] += relu(h[src] + e)
__global__ void msg_scatter(float* __restrict__ aggr, const float* __restrict__ h,
                            const float* __restrict__ e, const int* __restrict__ src,
                            const int* __restrict__ dst) {
  const long tot = (long)NE * 16;
  long i = (long)blockIdx.x * blockDim.x + threadIdx.x;
  const long gs = (long)gridDim.x * blockDim.x;
  for (; i < tot; i += gs) {
    const long ed = i >> 4;
    const int  c4 = (int)(i & 15) * 4;
    const float4 hv = *(const float4*)(h + (size_t)src[ed] * 64 + c4);
    const float4 ev = *(const float4*)(e + ed * 64 + c4);
    float* ap = aggr + (size_t)dst[ed] * 64 + c4;
    float v;
    v = hv.x + ev.x; if (v > 0.f) unsafeAtomicAdd(ap + 0, v);
    v = hv.y + ev.y; if (v > 0.f) unsafeAtomicAdd(ap + 1, v);
    v = hv.z + ev.z; if (v > 0.f) unsafeAtomicAdd(ap + 2, v);
    v = hv.w + ev.w; if (v > 0.f) unsafeAtomicAdd(ap + 3, v);
  }
}

// xl = (1+eps)*h + aggr
__global__ void combine_kernel(float* __restrict__ xl, const float* __restrict__ h,
                               const float* __restrict__ aggr, const float* __restrict__ eps_p) {
  const float k = 1.f + eps_p[0];
  const long tot = (long)NN * 64;
  long i = (long)blockIdx.x * blockDim.x + threadIdx.x;
  const long gs = (long)gridDim.x * blockDim.x;
  for (; i < tot; i += gs) xl[i] = k * h[i] + aggr[i];
}

// h = relu(bn(pre)) + px ; px = h   (bn scale/shift from stats buffer)
__global__ void residual_update(float* __restrict__ h, float* __restrict__ px,
                                const float* __restrict__ pre, const float* __restrict__ st) {
  const long tot = (long)NN * 64;
  long i = (long)blockIdx.x * blockDim.x + threadIdx.x;
  const long gs = (long)gridDim.x * blockDim.x;
  for (; i < tot; i += gs) {
    const int c = (int)(i & 63);
    float v = pre[i] * st[256 + c] + st[384 + c];
    v = (v > 0.f ? v : 0.f) + px[i];
    h[i] = v; px[i] = v;
  }
}

__global__ void pool_nodes(float* __restrict__ xg, const float* __restrict__ h,
                           const int* __restrict__ batch) {
  const long tot = (long)NN * 16;
  long i = (long)blockIdx.x * blockDim.x + threadIdx.x;
  const long gs = (long)gridDim.x * blockDim.x;
  for (; i < tot; i += gs) {
    const long n = i >> 4;
    const int  c4 = (int)(i & 15) * 4;
    const int  b = batch[n];
    const float4 v = *(const float4*)(h + n * 64 + c4);
    float* o = xg + (size_t)b * 64 + c4;
    unsafeAtomicAdd(o + 0, v.x); unsafeAtomicAdd(o + 1, v.y);
    unsafeAtomicAdd(o + 2, v.z); unsafeAtomicAdd(o + 3, v.w);
  }
}

__global__ void pool_edges(float* __restrict__ eg, const float* __restrict__ e,
                           const int* __restrict__ src, const int* __restrict__ batch) {
  const long tot = (long)NE * 16;
  long i = (long)blockIdx.x * blockDim.x + threadIdx.x;
  const long gs = (long)gridDim.x * blockDim.x;
  for (; i < tot; i += gs) {
    const long ed = i >> 4;
    const int  c4 = (int)(i & 15) * 4;
    const int  b = batch[src[ed]];
    const float4 v = *(const float4*)(e + ed * 64 + c4);
    float* o = eg + (size_t)b * 64 + c4;
    unsafeAtomicAdd(o + 0, v.x); unsafeAtomicAdd(o + 1, v.y);
    unsafeAtomicAdd(o + 2, v.z); unsafeAtomicAdd(o + 3, v.w);
  }
}

// ------------------------------------------------------------------- host
static inline unsigned gsz(long total) { return (unsigned)((total + 255) / 256); }

extern "C" void kernel_launch(void* const* d_in, const int* in_sizes, int n_in,
                              void* d_out, int out_size, void* d_ws, size_t ws_size,
                              hipStream_t stream) {
  (void)in_sizes; (void)out_size; (void)ws_size;

  // ---- d_in layout: JAX pytree flatten (dict keys alphabetical) ----
  // 0:batch 1:edge_attr 2:edge_index 3..59:convs 60..63:edge_enc
  // 64..67:input_enc 68..69:out1 70..73:out2 74..81:out_mlp last:x
  auto F = [&](int i) { return (const float*)d_in[i]; };
  const int*   batch = (const int*)d_in[0];
  const float* edge_attr = F(1);
  const int*   eidx = (const int*)d_in[2];
  const int*   src = eidx;
  const int*   dst = eidx + NE;
  const float* x = F(n_in - 1);
  const int PC = 3, PE = PC + 19 * 3, PI = PE + 4, PO1 = PI + 4, PO2 = PO1 + 2, PM = PO2 + 4;
  // per-conv: +0 eu_lin.W +1 eu_lin.b | eu_mlp0: +2 W +3 b +4 be +5 g
  // eu_mlp1: +6 W +7 b +8 be +9 g | +10 eps | gine_mlp0: +11..14 (W,b,be,g)
  // gine_mlp1: +15 W +16 b | norm: +17 be +18 g

  // ---- workspace layout (fp32 units) ----
  float* w  = (float*)d_ws;
  float* h  = w; w += (size_t)NN * 64;
  float* px = w; w += (size_t)NN * 64;
  float* xl = w; w += (size_t)NN * 64;
  float* ag = w; w += (size_t)NN * 64;
  float* n1 = w; w += (size_t)NN * 64;
  float* st = w; w += 512;                 // sum|sumsq|scale|shift
  half_t* whi = (half_t*)w;                // pre-split weights (max 128x128)
  half_t* wlo = whi + 128 * 128; w += 128 * 128;  // 32768 halves = 16384 floats
  float* xg = w; w += (size_t)NG * 64;
  float* eg = w; w += (size_t)NG * 64;
  float* z  = w; w += (size_t)NG * 128;
  float* z2 = w; w += (size_t)NG * 128;
  float* e  = w; w += (size_t)NE * 64;
  float* t1 = w; w += (size_t)NE * 128;
  float* t2 = w; w += (size_t)NE * 128;
  float* out = (float*)d_out;

  auto gemm = [&](int NT, int KT, const float* X, const float* W, const float* b,
                  float* Y, int M, int ld, int noff, int do_stats, const float* xf) {
    const int K = KT * 32, N = NT * 16;
    wsplit<<<(K * N + 255) / 256, 256, 0, stream>>>(W, whi, wlo, K, N);
    if (do_stats) zero_f32<<<1, 256, 0, stream>>>(st, 256);
    dim3 grid((M + 127) / 128), block(256);
    const size_t sh = (size_t)(2 * (128 + N) * (K + 8)) * sizeof(half_t) + 1024;
    if (NT == 8)      gemm_kernel<8, 4><<<grid, block, sh, stream>>>(X, whi, wlo, b, Y, st, xf, M, ld, noff, do_stats);
    else if (NT == 1) gemm_kernel<1, 4><<<grid, block, sh, stream>>>(X, whi, wlo, b, Y, st, xf, M, ld, noff, do_stats);
    else if (KT == 4) gemm_kernel<4, 4><<<grid, block, sh, stream>>>(X, whi, wlo, b, Y, st, xf, M, ld, noff, do_stats);
    else              gemm_kernel<4, 2><<<grid, block, sh, stream>>>(X, whi, wlo, b, Y, st, xf, M, ld, noff, do_stats);
  };
  auto fin = [&](int N, const float* g, const float* be, long M) {
    bn_finalize<<<1, 128, 0, stream>>>(st, g, be, 1.f / (float)M, N);
  };

  zero_f32<<<gsz((long)NN * 64), 256, 0, stream>>>(px, (long)NN * 64);

  // input_enc / edge_enc (Linear + BN + ReLU; h and e have many consumers -> materialize)
  gemm(4, 2, x, F(PI + 0), F(PI + 1), h, NN, 64, 0, 1, nullptr);
  fin(64, F(PI + 3), F(PI + 2), NN);
  bn_apply<<<gsz((long)NN * 16), 256, 0, stream>>>(h, st, NN, 64, 64);
  gemm(4, 2, edge_attr, F(PE + 0), F(PE + 1), e, NE, 64, 0, 1, nullptr);
  fin(64, F(PE + 3), F(PE + 2), NE);
  bn_apply<<<gsz((long)NE * 16), 256, 0, stream>>>(e, st, NE, 64, 64);

  for (int L = 0; L < 3; ++L) {
    const int B = PC + 19 * L;
    // EdgeUpdater
    gemm(4, 2, h, F(B + 0), F(B + 1), xl, NN, 64, 0, 0, nullptr);        // eu_lin
    concat_gather<<<gsz((long)NE * 32), 256, 0, stream>>>(t1, xl, e, src, dst);
    gemm(8, 4, t1, F(B + 2), F(B + 3), t2, NE, 128, 0, 1, nullptr);      // eu_mlp[0]
    fin(128, F(B + 5), F(B + 4), NE);                                    // BN fused into next GEMM
    gemm(4, 4, t2, F(B + 6), F(B + 7), e, NE, 64, 0, 1, st);             // eu_mlp[1] (+fused BN0)
    fin(64, F(B + 9), F(B + 8), NE);
    bn_apply<<<gsz((long)NE * 16), 256, 0, stream>>>(e, st, NE, 64, 64);
    // GINEConv
    zero_f32<<<gsz((long)NN * 64), 256, 0, stream>>>(ag, (long)NN * 64);
    msg_scatter<<<gsz((long)NE * 16), 256, 0, stream>>>(ag, h, e, src, dst);
    combine_kernel<<<gsz((long)NN * 64), 256, 0, stream>>>(xl, h, ag, F(B + 10));
    gemm(4, 2, xl, F(B + 11), F(B + 12), n1, NN, 64, 0, 1, nullptr);     // gine_mlp0
    fin(64, F(B + 14), F(B + 13), NN);                                   // fused into next GEMM
    gemm(4, 2, n1, F(B + 15), F(B + 16), ag, NN, 64, 0, 1, st);          // gine_mlp1 (+fused BN)
    fin(64, F(B + 18), F(B + 17), NN);                                   // "norm" BN
    residual_update<<<gsz((long)NN * 64), 256, 0, stream>>>(h, px, ag, st);
  }

  // pooling (segment-sum add)
  zero_f32<<<1, 256, 0, stream>>>(xg, (long)NG * 64);
  zero_f32<<<1, 256, 0, stream>>>(eg, (long)NG * 64);
  pool_nodes<<<gsz((long)NN * 16), 256, 0, stream>>>(xg, h, batch);
  pool_edges<<<gsz((long)NE * 16), 256, 0, stream>>>(eg, e, src, batch);

  // heads: z = [ out1(xg) , relu(bn(out2(eg))) ]
  gemm(4, 2, xg, F(PO1 + 0), F(PO1 + 1), z, NG, 128, 0, 0, nullptr);
  gemm(4, 2, eg, F(PO2 + 0), F(PO2 + 1), z, NG, 128, 64, 1, nullptr);
  fin(64, F(PO2 + 3), F(PO2 + 2), NG);
  bn_apply<<<gsz((long)NG * 16), 256, 0, stream>>>(z + 64, st, NG, 64, 128);

  // out_mlp
  gemm(8, 4, z, F(PM + 0), F(PM + 1), z2, NG, 128, 0, 1, nullptr);
  fin(128, F(PM + 3), F(PM + 2), NG);                                    // fused into next GEMM
  gemm(1, 4, z2, F(PM + 4), F(PM + 5), out, NG, 16, 0, 1, st);
  fin(16, F(PM + 7), F(PM + 6), NG);
  bn_apply<<<gsz((long)NG * 4), 256, 0, stream>>>(out, st, NG, 16, 16);
}
